// ConvNeXtParallelSSMBlock_13091060318286
// MI455X (gfx1250) — matrix operand order repaired and checked by reading, not compiled, not run on hardware
//
#include <hip/hip_runtime.h>
#include <hip/hip_bf16.h>
#include <math.h>

// Problem constants (from reference setup_inputs)
#define BB   8
#define HH   56
#define WW2  56
#define HW   3136      // 56*56
#define CC   192
#define HID  768
#define KSZ  7
#define TSTEPS 8
#define ROWS_PER_WG 64

typedef __attribute__((ext_vector_type(8)))  float  v8f;
typedef __attribute__((ext_vector_type(16))) __bf16 v16bf;
typedef __attribute__((ext_vector_type(8)))  __bf16 v8bf;

// ---------------------------------------------------------------------------
// Kernel 1: per-(bin,channel) combined frequency multiplier
//   M_f = dw_f * B_f * sum_{t=0..7} A_f^t   (A_stable = 0.9*tanh(A))
// ---------------------------------------------------------------------------
__global__ __launch_bounds__(256) void k_multiplier(
    const float* __restrict__ dw, const float* __restrict__ Ak,
    const float* __restrict__ Bk,
    float* __restrict__ Mre, float* __restrict__ Mim) {
  int idx = blockIdx.x * blockDim.x + threadIdx.x;
  if (idx >= HW * CC) return;
  int c  = idx / HW;
  int hw = idx - c * HW;
  int p = hw / WW2, q = hw - p * WW2;

  const float TWO_PI = 6.2831853071795864769f;
  float dwr = 0.f, dwi = 0.f, ar = 0.f, ai = 0.f, br = 0.f, bi = 0.f;
#pragma unroll
  for (int i = 0; i < KSZ; ++i) {
#pragma unroll
    for (int j = 0; j < KSZ; ++j) {
      float ph = (float)(p * (i - 3) + q * (j - 3));
      float ang = -TWO_PI * ph * (1.0f / 56.0f);
      float s, co;
      __sincosf(ang, &s, &co);
      int t = c * 49 + i * 7 + j;
      float kd = dw[t];
      float ka = 0.9f * tanhf(Ak[t]);
      float kb = Bk[t];
      dwr += kd * co; dwi += kd * s;
      ar  += ka * co; ai  += ka * s;
      br  += kb * co; bi  += kb * s;
    }
  }
  float sr = 1.f, si = 0.f, pr = 1.f, pi = 0.f;
#pragma unroll
  for (int t = 1; t < TSTEPS; ++t) {
    float nr = pr * ar - pi * ai;
    float ni = pr * ai + pi * ar;
    pr = nr; pi = ni;
    sr += pr; si += pi;
  }
  float t1r = dwr * br - dwi * bi;
  float t1i = dwr * bi + dwi * br;
  Mre[c * HW + hw] = t1r * sr - t1i * si;
  Mim[c * HW + hw] = t1r * si + t1i * sr;
}

// ---------------------------------------------------------------------------
// Kernel 2: weight pre-conversion f32 -> bf16 (once per launch)
// ---------------------------------------------------------------------------
__global__ __launch_bounds__(256) void k_cvtw(
    const float* __restrict__ W1, const float* __restrict__ W2,
    __bf16* __restrict__ W1b, __bf16* __restrict__ W2b) {
  int idx = blockIdx.x * blockDim.x + threadIdx.x;
  const int N1 = CC * HID;
  if (idx < N1)            W1b[idx]      = (__bf16)W1[idx];
  else if (idx < 2 * N1)   W2b[idx - N1] = (__bf16)W2[idx - N1];
}

// ---------------------------------------------------------------------------
// Kernel 3: spectral per-channel circular conv for one (b,c) image.
//   h = ifft2( M_f * fft2(x) ).real   (56-pt DFTs; 56 distinct twiddles,
//   index tracked incrementally mod 56; whole image resident in LDS)
// ---------------------------------------------------------------------------
__global__ __launch_bounds__(256) void k_specconv(
    const float* __restrict__ x,
    const float* __restrict__ Mre, const float* __restrict__ Mim,
    float* __restrict__ h) {
  __shared__ float twc[HH], tws[HH];
  __shared__ float Xr[HW], Xi[HW], Yr[HW], Yi[HW];

  int tid = threadIdx.x;
  int b = blockIdx.x / CC;
  int c = blockIdx.x - b * CC;

  if (tid < HH) {
    float ang = -6.2831853071795864769f * (float)tid * (1.0f / 56.0f);
    float s, co;
    __sincosf(ang, &s, &co);
    twc[tid] = co; tws[tid] = s;
  }
  for (int i = tid; i < HW; i += 256) {
    Xr[i] = x[(size_t)(b * HW + i) * CC + c];
    Xi[i] = 0.f;
  }
  __syncthreads();

  // pass 1: row DFT
  for (int i = tid; i < HW; i += 256) {
    int y = i / WW2, q = i - y * WW2;
    const float* xr = Xr + y * WW2;
    const float* xi = Xi + y * WW2;
    float sr = 0.f, si = 0.f;
    int m = 0;
    for (int n = 0; n < WW2; ++n) {
      float cr = twc[m], ci = tws[m];
      sr += xr[n] * cr - xi[n] * ci;
      si += xr[n] * ci + xi[n] * cr;
      m += q; if (m >= HH) m -= HH;
    }
    Yr[i] = sr; Yi[i] = si;
  }
  __syncthreads();

  // pass 2: col DFT + multiply by M[c]
  for (int i = tid; i < HW; i += 256) {
    int p = i / WW2, q = i - p * WW2;
    float sr = 0.f, si = 0.f;
    int m = 0;
    for (int y = 0; y < HH; ++y) {
      float cr = twc[m], ci = tws[m];
      float yr = Yr[y * WW2 + q], yi = Yi[y * WW2 + q];
      sr += yr * cr - yi * ci;
      si += yr * ci + yi * cr;
      m += p; if (m >= HH) m -= HH;
    }
    float mr = Mre[c * HW + i], mi = Mim[c * HW + i];
    Xr[i] = sr * mr - si * mi;
    Xi[i] = sr * mi + si * mr;
  }
  __syncthreads();

  // pass 3: inverse col DFT
  for (int i = tid; i < HW; i += 256) {
    int y = i / WW2, q = i - y * WW2;
    float sr = 0.f, si = 0.f;
    int m = 0;
    for (int p = 0; p < HH; ++p) {
      float cr = twc[m], ci = -tws[m];
      float xr = Xr[p * WW2 + q], xi = Xi[p * WW2 + q];
      sr += xr * cr - xi * ci;
      si += xr * ci + xi * cr;
      m += y; if (m >= HH) m -= HH;
    }
    Yr[i] = sr; Yi[i] = si;
  }
  __syncthreads();

  // pass 4: inverse row DFT, real part, scale 1/(56*56)
  for (int i = tid; i < HW; i += 256) {
    int y = i / WW2, n = i - y * WW2;
    const float* yr = Yr + y * WW2;
    const float* yi = Yi + y * WW2;
    float sr = 0.f;
    int m = 0;
    for (int q = 0; q < WW2; ++q) {
      float cr = twc[m], ci = -tws[m];
      sr += yr[q] * cr - yi[q] * ci;
      m += n; if (m >= HH) m -= HH;
    }
    h[(size_t)(b * HW + i) * CC + c] = sr * (1.0f / 3136.0f);
  }
}

// ---------------------------------------------------------------------------
// Kernel 4: fused LayerNorm -> MLP(GELU tanh) -> LayerScale -> residual.
// 64 rows per WG (4 M-tiles), 8 waves, v_wmma_f32_16x16x32_bf16.
// One B-fragment (bf16 weights) feeds 4 WMMAs (M-tile reuse).
// ---------------------------------------------------------------------------
__device__ __forceinline__ float gelu_tanh(float v) {
  float u = 0.7978845608028654f * (v + 0.044715f * v * v * v);
  return 0.5f * v * (1.0f + tanhf(u));
}

// A-fragment: 16xK LDS tile row-major, stride ld (elements).
// lanes 0-15: M=lane, K = {kb..kb+7, kb+16..kb+23}; lanes 16-31: kb += 8.
// Two aligned 16B LDS loads per lane.
__device__ __forceinline__ v16bf load_a_frag(const __bf16* base, int ld,
                                             int lane, int k0) {
  int row = lane & 15;
  int kb  = k0 + ((lane & 16) ? 8 : 0);
  const __bf16* rp = base + row * ld + kb;
  v8bf lo = *(const v8bf*)(rp);
  v8bf hi = *(const v8bf*)(rp + 16);
  return __builtin_shufflevector(lo, hi, 0, 1, 2, 3, 4, 5, 6, 7,
                                 8, 9, 10, 11, 12, 13, 14, 15);
}

// B-fragment: 32x16 tile of row-major bf16 weights [K][N].
// lanes 0-15: N=lane, K = {kb..kb+7, kb+16..kb+23}; lanes 16-31: kb += 8.
__device__ __forceinline__ v16bf load_b_frag(const __bf16* __restrict__ W,
                                             int ld, int lane, int k0, int n0) {
  int n  = n0 + (lane & 15);
  int kb = k0 + ((lane & 16) ? 8 : 0);
  v16bf r;
#pragma unroll
  for (int i = 0; i < 8; ++i) {
    r[i]     = W[(size_t)(kb + i) * ld + n];
    r[8 + i] = W[(size_t)(kb + 16 + i) * ld + n];
  }
  return r;
}

__global__ __launch_bounds__(256) void k_ln_mlp(
    const float* __restrict__ h, const float* __restrict__ x,
    const float* __restrict__ lns, const float* __restrict__ lnb,
    const __bf16* __restrict__ W1b, const float* __restrict__ b1,
    const __bf16* __restrict__ W2b, const float* __restrict__ b2,
    const float* __restrict__ gamma, float* __restrict__ out) {
  // dynamic LDS partition (all 32B aligned):
  //   HN  [64][CC]  bf16   24576 B
  //   ACT [64][HID] bf16   98304 B
  //   red1[256] red2[256] mu_s[64] rs_s[64] f32
  extern __shared__ char smem[];
  __bf16* HN  = (__bf16*)smem;
  __bf16* ACT = HN + ROWS_PER_WG * CC;
  float*  red1 = (float*)(ACT + ROWS_PER_WG * HID);
  float*  red2 = red1 + 256;
  float*  mu_s = red2 + 256;
  float*  rs_s = mu_s + ROWS_PER_WG;

  int tid = threadIdx.x;
  int rowBase = blockIdx.x * ROWS_PER_WG;

  // ---- LayerNorm stats: 4 threads per row, 64 rows ----
  {
    int r = tid >> 2, g = tid & 3;
    const float* hrow = h + (size_t)(rowBase + r) * CC;
    float s1 = 0.f, s2 = 0.f;
    for (int j = g; j < CC; j += 4) {
      float v = hrow[j];
      s1 += v; s2 += v * v;
    }
    red1[tid] = s1; red2[tid] = s2;
  }
  __syncthreads();
  {
    int r = tid >> 2, g = tid & 3;
    if (g == 0) {
      float a = red1[4 * r] + red1[4 * r + 1] + red1[4 * r + 2] + red1[4 * r + 3];
      float b = red2[4 * r] + red2[4 * r + 1] + red2[4 * r + 2] + red2[4 * r + 3];
      float mu  = a * (1.0f / CC);
      float var = b * (1.0f / CC) - mu * mu;
      mu_s[r] = mu;
      rs_s[r] = rsqrtf(var + 1e-6f);
    }
  }
  __syncthreads();

  // ---- normalize + affine -> bf16 LDS ----
  for (int i = tid; i < ROWS_PER_WG * CC; i += 256) {
    int rr = i / CC, jj = i - rr * CC;
    float v = (h[(size_t)(rowBase + rr) * CC + jj] - mu_s[rr]) * rs_s[rr]
              * lns[jj] + lnb[jj];
    HN[rr * CC + jj] = (__bf16)v;
  }
  __syncthreads();

  int wave = tid >> 5;
  int lane = tid & 31;
  int nl   = lane & 15;
  int mb   = (lane & 16) ? 8 : 0;

  // ---- GEMM1: [64x192] @ [192x768] + b1, GELU -> ACT (48 N-tiles / 8 waves)
  for (int t = wave; t < HID / 16; t += 8) {   // wave-uniform: EXEC stays full
    int n0 = t * 16;
    float bias = b1[n0 + nl];
    v8f acc[4];
#pragma unroll
    for (int mt = 0; mt < 4; ++mt)
#pragma unroll
      for (int rr = 0; rr < 8; ++rr) acc[mt][rr] = bias;
#pragma unroll
    for (int k0 = 0; k0 < CC; k0 += 32) {
      v16bf bb = load_b_frag(W1b, HID, lane, k0, n0);   // reused by 4 WMMAs
#pragma unroll
      for (int mt = 0; mt < 4; ++mt) {
        v16bf a = load_a_frag(HN + mt * 16 * CC, CC, lane, k0);
        acc[mt] = __builtin_amdgcn_wmma_f32_16x16x32_bf16(
            false, a, false, bb, (short)0, acc[mt], false, false);
      }
    }
    int col = n0 + nl;
#pragma unroll
    for (int mt = 0; mt < 4; ++mt)
#pragma unroll
      for (int rr = 0; rr < 8; ++rr)
        ACT[(mt * 16 + mb + rr) * HID + col] = (__bf16)gelu_tanh(acc[mt][rr]);
  }
  __syncthreads();

  // ---- GEMM2: [64x768] @ [768x192] + b2, *gamma + residual -> out
  for (int t = wave; t < CC / 16; t += 8) {    // wave-uniform: EXEC stays full
    int n0 = t * 16;
    int col = n0 + nl;
    float bias = b2[col];
    v8f acc[4];
#pragma unroll
    for (int mt = 0; mt < 4; ++mt)
#pragma unroll
      for (int rr = 0; rr < 8; ++rr) acc[mt][rr] = bias;
    for (int k0 = 0; k0 < HID; k0 += 32) {
      v16bf bb = load_b_frag(W2b, CC, lane, k0, n0);    // reused by 4 WMMAs
#pragma unroll
      for (int mt = 0; mt < 4; ++mt) {
        v16bf a = load_a_frag(ACT + mt * 16 * HID, HID, lane, k0);
        acc[mt] = __builtin_amdgcn_wmma_f32_16x16x32_bf16(
            false, a, false, bb, (short)0, acc[mt], false, false);
      }
    }
    float gm = gamma[col];
#pragma unroll
    for (int mt = 0; mt < 4; ++mt)
#pragma unroll
      for (int rr = 0; rr < 8; ++rr) {
        int row = rowBase + mt * 16 + mb + rr;
        out[(size_t)row * CC + col] = acc[mt][rr] * gm + x[(size_t)row * CC + col];
      }
  }
}

// ---------------------------------------------------------------------------
extern "C" void kernel_launch(void* const* d_in, const int* in_sizes, int n_in,
                              void* d_out, int out_size, void* d_ws,
                              size_t ws_size, hipStream_t stream) {
  (void)in_sizes; (void)n_in; (void)out_size; (void)ws_size;
  const float* x     = (const float*)d_in[0];
  const float* dw    = (const float*)d_in[1];
  const float* Ak    = (const float*)d_in[2];
  const float* Bk    = (const float*)d_in[3];
  const float* lns   = (const float*)d_in[4];
  const float* lnb   = (const float*)d_in[5];
  const float* W1    = (const float*)d_in[6];
  const float* b1    = (const float*)d_in[7];
  const float* W2    = (const float*)d_in[8];
  const float* b2    = (const float*)d_in[9];
  const float* gamma = (const float*)d_in[10];
  float* out = (float*)d_out;

  // workspace (all 32B aligned): Mre, Mim (HW*C f32), h (B*HW*C f32),
  // W1b, W2b (bf16)  => ~24.7 MB
  float*  Mre  = (float*)d_ws;
  float*  Mim  = Mre + (size_t)HW * CC;
  float*  hbuf = Mim + (size_t)HW * CC;
  __bf16* W1b  = (__bf16*)(hbuf + (size_t)BB * HW * CC);
  __bf16* W2b  = W1b + (size_t)CC * HID;

  k_multiplier<<<(HW * CC + 255) / 256, 256, 0, stream>>>(dw, Ak, Bk, Mre, Mim);
  k_cvtw<<<(2 * CC * HID + 255) / 256, 256, 0, stream>>>(W1, W2, W1b, W2b);
  k_specconv<<<BB * CC, 256, 0, stream>>>(x, Mre, Mim, hbuf);

  size_t smem = (size_t)ROWS_PER_WG * CC * sizeof(__bf16)      // HN
              + (size_t)ROWS_PER_WG * HID * sizeof(__bf16)     // ACT
              + (256 + 256 + ROWS_PER_WG + ROWS_PER_WG) * sizeof(float);
  k_ln_mlp<<<(BB * HW) / ROWS_PER_WG, 256, smem, stream>>>(
      hbuf, x, lns, lnb, W1b, b1, W2b, b2, gamma, out);
}
